// TextLSTM2_58420145160675
// MI455X (gfx1250) — matrix operand order repaired and checked by reading, not compiled, not run on hardware
//
#include <hip/hip_runtime.h>
#include <hip/hip_bf16.h>

#define SEQ   512
#define BATCH 256
#define EMB   256
#define HID   512
#define NBLK  64          // 64 blocks x 8 waves = 512 wave-tiles (16 m-tiles x 32 n-tiles)
#define TPB   256
#define WROW  (HID + 16)  // padded LDS row for Waa (32B-aligned stride, bank-spread)
#define GROW  (EMB + 16)  // padded LDS row for gate weights

typedef __attribute__((ext_vector_type(16))) __bf16 v16bf;
typedef __attribute__((ext_vector_type(8)))  __bf16 v8bf;
typedef __attribute__((ext_vector_type(8)))  float  v8f;

__device__ __forceinline__ unsigned short f32_to_bf16_rne(float f) {
  unsigned u = __float_as_uint(f);
  u += 0x7FFFu + ((u >> 16) & 1u);        // round-to-nearest-even
  return (unsigned short)(u >> 16);
}

// ---------------- elementwise f32 -> bf16 conversion ----------------
__global__ void cvt_bf16_kernel(const float* __restrict__ src,
                                unsigned short* __restrict__ dst, int n) {
  int i = blockIdx.x * blockDim.x + threadIdx.x;
  if (i < n) dst[i] = f32_to_bf16_rne(src[i]);
}

// ---------------- zero h state and barrier counters ----------------
__global__ void init_state_kernel(unsigned short* __restrict__ hbf,
                                  unsigned* __restrict__ cnt) {
  int i = blockIdx.x * blockDim.x + threadIdx.x;
  if (i < BATCH * HID) hbf[i] = 0;
  if (i < SEQ) cnt[i] = 0u;
}

// 16x32 bf16 A-fragment (contiguous k order, not the ISA table order):
// lane half h: elements e=0..15 -> K = kb + 16*h + e (contiguous, matches B fragments)
__device__ __forceinline__ v16bf load_a_frag(const unsigned short* __restrict__ row,
                                             int kb, int half) {
  v8bf lo = *(const v8bf*)(row + kb + 16 * half);
  v8bf hi = *(const v8bf*)(row + kb + 16 * half + 8);
  return __builtin_shufflevector(lo, hi, 0,1,2,3,4,5,6,7,8,9,10,11,12,13,14,15);
}

__device__ __forceinline__ float sigmoid_f(float x) {
  return 1.0f / (1.0f + __expf(-x));
}
__device__ __forceinline__ float tanh_f(float x) {
  float e2 = __expf(2.0f * x);
  return (e2 - 1.0f) / (e2 + 1.0f);
}

// ---------------- persistent recurrence kernel ----------------
__global__ void __launch_bounds__(TPB)
lstm_persistent_kernel(const unsigned short* __restrict__ Xb,    // [SEQ][BATCH][EMB] bf16
                       const unsigned short* __restrict__ Wfb,   // [HID][EMB] bf16
                       const unsigned short* __restrict__ Wib,
                       const unsigned short* __restrict__ Wob,
                       const unsigned short* __restrict__ Wab,
                       const unsigned short* __restrict__ Waab,  // [HID][HID] bf16
                       const float* __restrict__ bfp, const float* __restrict__ bip,
                       const float* __restrict__ bop, const float* __restrict__ bap,
                       unsigned short* __restrict__ hbf,         // [BATCH][HID] bf16 state
                       unsigned* __restrict__ cnt,               // [SEQ] barrier counters
                       float* __restrict__ out) {                // [BATCH][HID] f32
  // All 8 waves of a block share one n-tile: tn = blockIdx.x >> 1.
  const int wave = threadIdx.x >> 5;
  const int lane = threadIdx.x & 31;
  const int n0   = (blockIdx.x >> 1) * 16;                 // hidden tile (block-uniform)
  const int m0   = ((blockIdx.x & 1) * 8 + wave) * 16;     // batch tile (per wave)
  const int half = lane >> 4;
  const int sub  = lane & 15;

  // ---- stage this block's weight columns into LDS (once; reused 512 steps) ----
  __shared__ __align__(32) unsigned short ldsW[16 * WROW + 4 * 16 * GROW];
  unsigned short* lwaa = ldsW;                 // [16][WROW]
  unsigned short* lwg  = ldsW + 16 * WROW;     // [4][16][GROW]
  for (int i = threadIdx.x; i < 16 * HID; i += TPB) {
    int r = i / HID, k = i % HID;
    lwaa[r * WROW + k] = Waab[(size_t)(n0 + r) * HID + k];
  }
  {
    const unsigned short* gsrc[4] = {Wfb, Wib, Wob, Wab};
    #pragma unroll
    for (int g = 0; g < 4; ++g)
      for (int i = threadIdx.x; i < 16 * EMB; i += TPB) {
        int r = i / EMB, k = i % EMB;
        lwg[(g * 16 + r) * GROW + k] = gsrc[g][(size_t)(n0 + r) * EMB + k];
      }
  }
  __syncthreads();

  // Per-lane B-fragment rows in LDS (N = n0 + sub).
  const unsigned short* lra = lwaa + sub * WROW;
  const unsigned short* lrf = lwg + (0 * 16 + sub) * GROW;
  const unsigned short* lri = lwg + (1 * 16 + sub) * GROW;
  const unsigned short* lro = lwg + (2 * 16 + sub) * GROW;
  const unsigned short* lrx = lwg + (3 * 16 + sub) * GROW;

  // Per-lane biases: C/D layout puts N = lane%16 for every accumulator VGPR.
  const float bfv = bfp[n0 + sub], biv = bip[n0 + sub];
  const float bov = bop[n0 + sub], bav = bap[n0 + sub];

  const unsigned short* hrow = hbf + (size_t)(m0 + sub) * HID;  // A rows (h)

  // Cell state lives in registers: this wave exclusively owns tile (m0,n0).
  float creg[8];
  #pragma unroll
  for (int r = 0; r < 8; ++r) creg[r] = 0.0f;

  for (int t = 0; t < SEQ; ++t) {
    // ---- ah = h @ Waa^T (K = 512 -> 16 WMMAs, 2 interleaved accumulators) ----
    v8f ah0 = {}, ah1 = {};
    #pragma unroll 2
    for (int kb = 0; kb < HID; kb += 64) {
      v16bf a0 = load_a_frag(hrow, kb, half);
      v16bf a1 = load_a_frag(hrow, kb + 32, half);
      v16bf b0 = *(const v16bf*)(lra + kb + 16 * half);
      v16bf b1 = *(const v16bf*)(lra + kb + 32 + 16 * half);
      ah0 = __builtin_amdgcn_wmma_f32_16x16x32_bf16(false, a0, false, b0, (short)0, ah0, false, false);
      ah1 = __builtin_amdgcn_wmma_f32_16x16x32_bf16(false, a1, false, b1, (short)0, ah1, false, false);
    }
    v8f ah = ah0 + ah1;

    // ---- four x-projections (K = 256 -> 8 k-blocks x 4 gates, A shared) ----
    v8f af = {}, ai = {}, ao = {}, aa = {};
    const unsigned short* xrow = Xb + ((size_t)t * BATCH + m0 + sub) * EMB;
    #pragma unroll 2
    for (int kb = 0; kb < EMB; kb += 32) {
      v16bf a  = load_a_frag(xrow, kb, half);
      v16bf b0 = *(const v16bf*)(lrf + kb + 16 * half);
      v16bf b1 = *(const v16bf*)(lri + kb + 16 * half);
      v16bf b2 = *(const v16bf*)(lro + kb + 16 * half);
      v16bf b3 = *(const v16bf*)(lrx + kb + 16 * half);
      af = __builtin_amdgcn_wmma_f32_16x16x32_bf16(false, a, false, b0, (short)0, af, false, false);
      ai = __builtin_amdgcn_wmma_f32_16x16x32_bf16(false, a, false, b1, (short)0, ai, false, false);
      ao = __builtin_amdgcn_wmma_f32_16x16x32_bf16(false, a, false, b2, (short)0, ao, false, false);
      aa = __builtin_amdgcn_wmma_f32_16x16x32_bf16(false, a, false, b3, (short)0, aa, false, false);
    }

    // ---- elementwise gates: VGPR r holds M = r + 8*half, N = sub ----
    #pragma unroll
    for (int r = 0; r < 8; ++r) {
      const int m   = m0 + r + 8 * half;
      const size_t idx = (size_t)m * HID + n0 + sub;
      float s  = ah[r];
      float ft = sigmoid_f(af[r] + s + 2.0f * bfv);   // bias added twice (faithful)
      float it = sigmoid_f(ai[r] + s + 2.0f * biv);
      float ot = sigmoid_f(ao[r] + s + 2.0f * bov);
      float cp = tanh_f  (aa[r] + s + bav);
      float cn = ft * creg[r] + it * cp;
      float h  = ot * tanh_f(cn);
      creg[r]  = cn;                                   // cell state stays in registers
      hbf[idx] = f32_to_bf16_rne(h);
      if (t == SEQ - 1) out[idx] = h;
    }

    // ---- device-wide barrier so step t+1 sees all h updates ----
    if (t + 1 < SEQ) {
      __threadfence();           // release our h stores to device scope
      __syncthreads();
      if (threadIdx.x == 0) {
        atomicAdd(&cnt[t], 1u);
        while (__hip_atomic_load(&cnt[t], __ATOMIC_ACQUIRE,
                                 __HIP_MEMORY_SCOPE_AGENT) < (unsigned)NBLK) {
          __builtin_amdgcn_s_sleep(2);
        }
      }
      __syncthreads();
      __threadfence();           // acquire
    }
  }
}

extern "C" void kernel_launch(void* const* d_in, const int* in_sizes, int n_in,
                              void* d_out, int out_size, void* d_ws, size_t ws_size,
                              hipStream_t stream) {
  (void)in_sizes; (void)n_in; (void)out_size; (void)ws_size;
  const float* X   = (const float*)d_in[0];
  const float* Wfx = (const float*)d_in[1];
  const float* Wix = (const float*)d_in[2];
  const float* Wox = (const float*)d_in[3];
  const float* Wax = (const float*)d_in[4];
  const float* Waa = (const float*)d_in[5];
  const float* bf  = (const float*)d_in[6];
  const float* bi  = (const float*)d_in[7];
  const float* bo  = (const float*)d_in[8];
  const float* ba  = (const float*)d_in[9];
  float* out = (float*)d_out;

  // workspace carve-up (all offsets 32B-aligned)
  unsigned short* Xb   = (unsigned short*)d_ws;                      // 512*256*256 bf16
  unsigned short* Wfb  = Xb  + (size_t)SEQ * BATCH * EMB;
  unsigned short* Wib  = Wfb + (size_t)HID * EMB;
  unsigned short* Wob  = Wib + (size_t)HID * EMB;
  unsigned short* Wab  = Wob + (size_t)HID * EMB;
  unsigned short* Waab = Wab + (size_t)HID * EMB;                    // 512*512 bf16
  unsigned short* hbf  = Waab + (size_t)HID * HID;                   // 256*512 bf16
  unsigned*       cnt  = (unsigned*)(hbf + (size_t)BATCH * HID);     // 512 u32

  const int nX = SEQ * BATCH * EMB;
  cvt_bf16_kernel<<<(nX + 255) / 256, 256, 0, stream>>>(X, Xb, nX);
  cvt_bf16_kernel<<<(HID * EMB + 255) / 256, 256, 0, stream>>>(Wfx, Wfb, HID * EMB);
  cvt_bf16_kernel<<<(HID * EMB + 255) / 256, 256, 0, stream>>>(Wix, Wib, HID * EMB);
  cvt_bf16_kernel<<<(HID * EMB + 255) / 256, 256, 0, stream>>>(Wox, Wob, HID * EMB);
  cvt_bf16_kernel<<<(HID * EMB + 255) / 256, 256, 0, stream>>>(Wax, Wab, HID * EMB);
  cvt_bf16_kernel<<<(HID * HID + 255) / 256, 256, 0, stream>>>(Waa, Waab, HID * HID);
  init_state_kernel<<<(BATCH * HID + 255) / 256, 256, 0, stream>>>(hbf, cnt);

  lstm_persistent_kernel<<<NBLK, TPB, 0, stream>>>(
      Xb, Wfb, Wib, Wob, Wab, Waab, bf, bi, bo, ba, hbf, cnt, out);
}
